// MultiheadAttentionWithHook_75316546503188
// MI455X (gfx1250) — matrix operand order repaired and checked by reading, not compiled, not run on hardware
//
#include <hip/hip_runtime.h>
#include <hip/hip_bf16.h>

// ---------------------------------------------------------------------------
// MHA forward for MI455X (gfx1250, wave32, WMMA).
// ~137 GFLOP vs ~150MB minimal traffic -> matrix-unit bound (>10x) at
// 23.3 TB/s. All GEMMs/attention run on v_wmma_f32_16x16x32_bf16 with f32
// accumulation. fp32->bf16 conversion is hoisted into one bandwidth-bound
// pass so GEMM inner loops are pure b128 copy + WMMA (no per-element VALU).
// ---------------------------------------------------------------------------

#define B_  4
#define S_  2048
#define E_  1024
#define H_  16
#define D_  64

typedef __attribute__((ext_vector_type(16))) __bf16        v16bf;
typedef __attribute__((ext_vector_type(8)))  float         v8f;
typedef __attribute__((ext_vector_type(4)))  unsigned int  u32x4;
typedef __attribute__((ext_vector_type(4)))  float         f32x4;

union FragU {
  struct { u32x4 lo, hi; } u;
  v16bf v;
};

__device__ __forceinline__ unsigned int pack_bf16x2(float a, float b) {
#if __has_builtin(__builtin_amdgcn_cvt_pk_bf16_f32)
  auto r = __builtin_amdgcn_cvt_pk_bf16_f32(a, b);   // v_cvt_pk_bf16_f32
  return __builtin_bit_cast(unsigned int, r);
#else
  typedef __attribute__((ext_vector_type(2))) __bf16 v2bf;
  v2bf v; v.x = (__bf16)a; v.y = (__bf16)b;          // let backend pick cvt
  return __builtin_bit_cast(unsigned int, v);
#endif
}

// A-operand fragment, 16x32 bf16: lane m = lane&15, half h = lane>>4.
// dwords 0..3 hold K = h*8 + [0..7], dwords 4..7 hold K = 16 + h*8 + [0..7].
__device__ __forceinline__ v16bf lds_frag_a(const unsigned short* base, int row,
                                            int ld, int k0) {
  const int lane = threadIdx.x & 31;
  const int m = lane & 15, hf = lane >> 4;
  const unsigned short* p = base + (row + m) * ld + k0 + hf * 8;
  FragU f;
  f.u.lo = *(const u32x4*)(p);
  f.u.hi = *(const u32x4*)(p + 16);
  return f.v;
}
// B-operand fragment, 32x16 bf16 from [n][k] row-major LDS:
// lane n = lane&15, half h = lane>>4; K = h*8 + [0..7], then K = 16 + h*8 + [0..7].
__device__ __forceinline__ v16bf lds_frag_b(const unsigned short* base, int row,
                                            int ld, int k0) {
  const int lane = threadIdx.x & 31;
  const int n = lane & 15, hf = lane >> 4;
  const unsigned short* p = base + (row + n) * ld + k0 + hf * 8;
  FragU f;
  f.u.lo = *(const u32x4*)(p);
  f.u.hi = *(const u32x4*)(p + 16);
  return f.v;
}

#define WMMA_BF16(a, b, c) \
  __builtin_amdgcn_wmma_f32_16x16x32_bf16(false, (a), false, (b), (short)0, (c), false, false)

// ---------------------------------------------------------------------------
// Elementwise fp32 -> bf16 (8 elems/thread, b128 in / b128 out).
// ---------------------------------------------------------------------------
__global__ __launch_bounds__(256) void cvt_bf16_kernel(
    const float* __restrict__ src, unsigned short* __restrict__ dst, int n8) {
  int i = blockIdx.x * 256 + threadIdx.x;
  if (i >= n8) return;
  const f32x4* s = (const f32x4*)src + 2 * (size_t)i;
  f32x4 a = s[0], b = s[1];
  u32x4 u;
  u.x = pack_bf16x2(a.x, a.y);  u.y = pack_bf16x2(a.z, a.w);
  u.z = pack_bf16x2(b.x, b.y);  u.w = pack_bf16x2(b.z, b.w);
  ((u32x4*)dst)[i] = u;
}

// ---------------------------------------------------------------------------
// GEMM config: 256 thr (8 waves), block tile 128x128, k-step 64.
// Wave tile 64x32: wm = (wid>>2)*64, wn = (wid&3)*32 -> 16 WMMA per k-step.
// LDS rows padded to 72 shorts (144B = 9*16B): conflict-free, 16B aligned.
// Register-pipelined double buffer: next tile's global_load_b128 issued
// before the WMMAs, stored to the other LDS buffer after them.
// ---------------------------------------------------------------------------
#define LDG 72

__device__ __forceinline__ void gemm_mma_step64(const unsigned short* AsS,
                                                const unsigned short* WsS,
                                                int wm, int wn, v8f acc[4][2]) {
#pragma unroll
  for (int ks = 0; ks < 2; ++ks) {
    v16bf bfm[2];
#pragma unroll
    for (int ni = 0; ni < 2; ++ni) bfm[ni] = lds_frag_b(WsS, wn + ni*16, LDG, ks*32);
#pragma unroll
    for (int mi = 0; mi < 4; ++mi) {
      v16bf a = lds_frag_a(AsS, wm + mi*16, LDG, ks*32);
#pragma unroll
      for (int ni = 0; ni < 2; ++ni)
        acc[mi][ni] = WMMA_BF16(a, bfm[ni], acc[mi][ni]);
    }
  }
}

// ---------------------------------------------------------------------------
// Projection GEMM: C[8192,1024] = A(bf16) @ W(bf16)^T -> bf16 [B,H,S,D].
// ---------------------------------------------------------------------------
__global__ __launch_bounds__(256) void proj_gemm_kernel(
    const unsigned short* __restrict__ A, const unsigned short* __restrict__ W,
    unsigned short* __restrict__ out_bhsd) {
  __shared__ unsigned short As[2][128][LDG];
  __shared__ unsigned short Ws[2][128][LDG];

  const int tid  = threadIdx.x;
  const int mb   = blockIdx.y * 128;
  const int nb   = blockIdx.x * 128;
  const int wid  = tid >> 5;
  const int lane = tid & 31;
  const int nl = lane & 15, hf = lane >> 4;
  const int wm = (wid >> 2) * 64;
  const int wn = (wid & 3) * 32;
  const int srow = tid >> 1, shalf = tid & 1;   // stage 32 shorts per thread

  const v8f vzero = {0.f,0.f,0.f,0.f,0.f,0.f,0.f,0.f};
  v8f acc[4][2];
#pragma unroll
  for (int i = 0; i < 4; ++i)
#pragma unroll
    for (int j = 0; j < 2; ++j) acc[i][j] = vzero;

  const unsigned short* ga = A + (size_t)(mb + srow) * E_ + shalf * 32;
  const unsigned short* gw = W + (size_t)(nb + srow) * E_ + shalf * 32;

  {  // prologue: stage k-block 0 into buffer 0
#pragma unroll
    for (int j = 0; j < 4; ++j)
      *(u32x4*)&As[0][srow][shalf*32 + j*8] = ((const u32x4*)ga)[j];
#pragma unroll
    for (int j = 0; j < 4; ++j)
      *(u32x4*)&Ws[0][srow][shalf*32 + j*8] = ((const u32x4*)gw)[j];
  }
  __syncthreads();

  for (int kb = 0; kb < E_ / 64; ++kb) {
    const int cur = kb & 1;
    const bool more = (kb + 1) < (E_ / 64);
    u32x4 ra[4] = {}, rw[4] = {};
    if (more) {  // issue next-tile global loads before compute
      const int k0n = (kb + 1) * 64;
#pragma unroll
      for (int j = 0; j < 4; ++j) ra[j] = ((const u32x4*)(ga + k0n))[j];
#pragma unroll
      for (int j = 0; j < 4; ++j) rw[j] = ((const u32x4*)(gw + k0n))[j];
      if (kb + 2 < E_ / 64) {
        __builtin_prefetch(ga + k0n + 64, 0, 1);   // global_prefetch_b8
        __builtin_prefetch(gw + k0n + 64, 0, 1);
      }
    }
    gemm_mma_step64(&As[cur][0][0], &Ws[cur][0][0], wm, wn, acc);
    if (more) {
      const int nxt = cur ^ 1;
#pragma unroll
      for (int j = 0; j < 4; ++j) *(u32x4*)&As[nxt][srow][shalf*32 + j*8] = ra[j];
#pragma unroll
      for (int j = 0; j < 4; ++j) *(u32x4*)&Ws[nxt][srow][shalf*32 + j*8] = rw[j];
    }
    __syncthreads();
  }

  // epilogue: bf16 scatter into [B,H,S,D]; C layout: M = r + 8*hf, N = nl.
#pragma unroll
  for (int mi = 0; mi < 4; ++mi)
#pragma unroll
    for (int r = 0; r < 8; ++r) {
      int m = mb + wm + mi*16 + r + 8*hf;
      int n0 = nb + wn + nl;
      int n1 = n0 + 16;
      int bb = m >> 11, s = m & (S_ - 1);
      unsigned int pk = pack_bf16x2(acc[mi][0][r], acc[mi][1][r]);
      out_bhsd[(((size_t)(bb*H_ + (n0 >> 6))) * S_ + s) * D_ + (n0 & (D_-1))] =
          (unsigned short)pk;
      out_bhsd[(((size_t)(bb*H_ + (n1 >> 6))) * S_ + s) * D_ + (n1 & (D_-1))] =
          (unsigned short)(pk >> 16);
    }
}

// ---------------------------------------------------------------------------
// Output GEMM: out(fp32)[8192,1024] = ctx(bf16) @ wfc(bf16)^T.
// ---------------------------------------------------------------------------
__global__ __launch_bounds__(256) void out_gemm_kernel(
    const unsigned short* __restrict__ A, const unsigned short* __restrict__ W,
    float* __restrict__ out) {
  __shared__ unsigned short As[2][128][LDG];
  __shared__ unsigned short Ws[2][128][LDG];

  const int tid  = threadIdx.x;
  const int mb   = blockIdx.y * 128;
  const int nb   = blockIdx.x * 128;
  const int wid  = tid >> 5;
  const int lane = tid & 31;
  const int nl = lane & 15, hf = lane >> 4;
  const int wm = (wid >> 2) * 64;
  const int wn = (wid & 3) * 32;
  const int srow = tid >> 1, shalf = tid & 1;

  const v8f vzero = {0.f,0.f,0.f,0.f,0.f,0.f,0.f,0.f};
  v8f acc[4][2];
#pragma unroll
  for (int i = 0; i < 4; ++i)
#pragma unroll
    for (int j = 0; j < 2; ++j) acc[i][j] = vzero;

  const unsigned short* ga = A + (size_t)(mb + srow) * E_ + shalf * 32;
  const unsigned short* gw = W + (size_t)(nb + srow) * E_ + shalf * 32;

  {
#pragma unroll
    for (int j = 0; j < 4; ++j)
      *(u32x4*)&As[0][srow][shalf*32 + j*8] = ((const u32x4*)ga)[j];
#pragma unroll
    for (int j = 0; j < 4; ++j)
      *(u32x4*)&Ws[0][srow][shalf*32 + j*8] = ((const u32x4*)gw)[j];
  }
  __syncthreads();

  for (int kb = 0; kb < E_ / 64; ++kb) {
    const int cur = kb & 1;
    const bool more = (kb + 1) < (E_ / 64);
    u32x4 ra[4] = {}, rw[4] = {};
    if (more) {
      const int k0n = (kb + 1) * 64;
#pragma unroll
      for (int j = 0; j < 4; ++j) ra[j] = ((const u32x4*)(ga + k0n))[j];
#pragma unroll
      for (int j = 0; j < 4; ++j) rw[j] = ((const u32x4*)(gw + k0n))[j];
      if (kb + 2 < E_ / 64) {
        __builtin_prefetch(ga + k0n + 64, 0, 1);
        __builtin_prefetch(gw + k0n + 64, 0, 1);
      }
    }
    gemm_mma_step64(&As[cur][0][0], &Ws[cur][0][0], wm, wn, acc);
    if (more) {
      const int nxt = cur ^ 1;
#pragma unroll
      for (int j = 0; j < 4; ++j) *(u32x4*)&As[nxt][srow][shalf*32 + j*8] = ra[j];
#pragma unroll
      for (int j = 0; j < 4; ++j) *(u32x4*)&Ws[nxt][srow][shalf*32 + j*8] = rw[j];
    }
    __syncthreads();
  }

#pragma unroll
  for (int mi = 0; mi < 4; ++mi)
#pragma unroll
    for (int r = 0; r < 8; ++r) {
      int m = mb + wm + mi*16 + r + 8*hf;
      int n0 = nb + wn + nl;
      out[(size_t)m * E_ + n0]      = acc[mi][0][r];
      out[(size_t)m * E_ + n0 + 16] = acc[mi][1][r];
    }
}

// ---------------------------------------------------------------------------
// Flash attention: block = 128 thr (4 waves) = 64 query rows of one (b,h).
// Each wave: 16 q-rows, Q A-fragments resident in VGPRs for the whole K loop.
// Per 64-key block: 8 WMMA (QK^T) + online softmax + 8 WMMA (PV).
// K/V tiles register-pipelined into double-buffered padded LDS.
// ---------------------------------------------------------------------------
#define LDA 72   // padded row stride (shorts): 144B = 9*16B, conflict-free

__global__ __launch_bounds__(128) void attn_kernel(
    const unsigned short* __restrict__ qws, const unsigned short* __restrict__ kws,
    const unsigned short* __restrict__ vws, const unsigned char* __restrict__ mask,
    unsigned short* __restrict__ ctx) {
  __shared__ unsigned short Ks[2][64][LDA];     // [key][d]
  __shared__ unsigned short Vt[2][64][LDA];     // [d][key]  (transposed V)
  __shared__ unsigned short Ps[4][16][LDA];     // per-wave P tile, bf16

  const int tid  = threadIdx.x;
  const int wid  = tid >> 5;
  const int lane = tid & 31;
  const int nl = lane & 15, hf = lane >> 4;

  const int bq   = blockIdx.z;
  const int hd   = blockIdx.y;
  const int qblk = blockIdx.x * 64;

  const size_t head_off = ((size_t)(bq * H_ + hd)) * S_ * D_;
  const unsigned short* qbp = qws + head_off;
  const unsigned short* kbp = kws + head_off;
  const unsigned short* vbp = vws + head_off;
  const unsigned char*  mp  = mask + (size_t)bq * S_;

  // Q A-fragments: 16 rows x 64 d -> two k-steps of 32.
  v16bf a_q[2];
  {
    const unsigned short* qp = qbp + (size_t)(qblk + wid*16 + nl) * D_;
#pragma unroll
    for (int ks = 0; ks < 2; ++ks) {
      FragU f;
      f.u.lo = *(const u32x4*)(qp + ks*32 + hf*8);
      f.u.hi = *(const u32x4*)(qp + ks*32 + 16 + hf*8);
      a_q[ks] = f.v;
    }
  }

  const v8f vzero = {0.f,0.f,0.f,0.f,0.f,0.f,0.f,0.f};
  v8f acco[4];
#pragma unroll
  for (int nd = 0; nd < 4; ++nd) acco[nd] = vzero;
  float mrow[8], lrow[8];
#pragma unroll
  for (int r = 0; r < 8; ++r) { mrow[r] = -1e30f; lrow[r] = 0.f; }

  const float SCL = 0.18033688011112042f;  // log2(e) / sqrt(D)
  const int row = tid >> 1, hsel = tid & 1;
  const unsigned short* kg = kbp + (size_t)row * D_ + hsel * 32;
  const unsigned short* vg = vbp + (size_t)row * D_ + hsel * 32;

  {  // prologue: stage key block 0 into buffer 0
    u32x4 rk[4], rv[4];
#pragma unroll
    for (int j = 0; j < 4; ++j) { rk[j] = ((const u32x4*)kg)[j]; rv[j] = ((const u32x4*)vg)[j]; }
#pragma unroll
    for (int j = 0; j < 4; ++j) *(u32x4*)&Ks[0][row][hsel*32 + j*8] = rk[j];
#pragma unroll
    for (int j = 0; j < 4; ++j) {
      u32x4 u = rv[j];
      int d0 = hsel*32 + j*8;
      Vt[0][d0+0][row] = (unsigned short)(u.x);
      Vt[0][d0+1][row] = (unsigned short)(u.x >> 16);
      Vt[0][d0+2][row] = (unsigned short)(u.y);
      Vt[0][d0+3][row] = (unsigned short)(u.y >> 16);
      Vt[0][d0+4][row] = (unsigned short)(u.z);
      Vt[0][d0+5][row] = (unsigned short)(u.z >> 16);
      Vt[0][d0+6][row] = (unsigned short)(u.w);
      Vt[0][d0+7][row] = (unsigned short)(u.w >> 16);
    }
  }
  __syncthreads();

  for (int kb = 0; kb < S_ / 64; ++kb) {
    const int cur  = kb & 1;
    const int key0 = kb * 64;
    const bool more = (kb + 1) < (S_ / 64);

    // 1) issue next K/V tile global loads early
    u32x4 rk[4] = {}, rv[4] = {};
    if (more) {
      const size_t goff = (size_t)(key0 + 64) * D_;
#pragma unroll
      for (int j = 0; j < 4; ++j) { rk[j] = ((const u32x4*)(kg + goff))[j]; rv[j] = ((const u32x4*)(vg + goff))[j]; }
      if (kb + 2 < S_ / 64) {
        __builtin_prefetch(kg + goff + 64 * D_, 0, 0);  // global_prefetch_b8
        __builtin_prefetch(vg + goff + 64 * D_, 0, 0);
      }
    }

    // 2) scores S = Q @ K^T
    v8f accs[4];
#pragma unroll
    for (int t = 0; t < 4; ++t) accs[t] = vzero;
#pragma unroll
    for (int ks = 0; ks < 2; ++ks)
#pragma unroll
      for (int t = 0; t < 4; ++t) {
        v16bf bk = lds_frag_b(&Ks[cur][0][0], t*16, LDA, ks*32);
        accs[t] = WMMA_BF16(a_q[ks], bk, accs[t]);
      }

    // key mask (True => -inf)
#pragma unroll
    for (int t = 0; t < 4; ++t) {
      float mv = mp[key0 + t*16 + nl] ? -1e30f : 0.f;
#pragma unroll
      for (int r = 0; r < 8; ++r) accs[t][r] += mv;
    }

    // online softmax: row max/sum across 16 lanes per half-wave
#pragma unroll
    for (int r = 0; r < 8; ++r) {
      float mx = fmaxf(fmaxf(accs[0][r], accs[1][r]),
                       fmaxf(accs[2][r], accs[3][r]));
      mx = fmaxf(mx, __shfl_xor(mx, 1, 16));
      mx = fmaxf(mx, __shfl_xor(mx, 2, 16));
      mx = fmaxf(mx, __shfl_xor(mx, 4, 16));
      mx = fmaxf(mx, __shfl_xor(mx, 8, 16));
      float mn  = fmaxf(mrow[r], mx);
      float fac = exp2f((mrow[r] - mn) * SCL);
      mrow[r] = mn;
      lrow[r] *= fac;
#pragma unroll
      for (int nd = 0; nd < 4; ++nd) acco[nd][r] *= fac;
    }
#pragma unroll
    for (int r = 0; r < 8; ++r) {
      float ps = 0.f;
#pragma unroll
      for (int t = 0; t < 4; ++t) {
        float p = exp2f((accs[t][r] - mrow[r]) * SCL);
        accs[t][r] = p;
        ps += p;
      }
      ps += __shfl_xor(ps, 1, 16);
      ps += __shfl_xor(ps, 2, 16);
      ps += __shfl_xor(ps, 4, 16);
      ps += __shfl_xor(ps, 8, 16);
      lrow[r] += ps;
    }

    // 3) P -> per-wave LDS tile (paired bf16 cvt)
#pragma unroll
    for (int r = 0; r < 8; ++r) {
      unsigned int p01 = pack_bf16x2(accs[0][r], accs[1][r]);
      unsigned int p23 = pack_bf16x2(accs[2][r], accs[3][r]);
      int rr = r + 8*hf;
      Ps[wid][rr][nl]      = (unsigned short)p01;
      Ps[wid][rr][16 + nl] = (unsigned short)(p01 >> 16);
      Ps[wid][rr][32 + nl] = (unsigned short)p23;
      Ps[wid][rr][48 + nl] = (unsigned short)(p23 >> 16);
    }
    // wave-local LDS fence: P stores above feed cross-lane frag loads below;
    // staging stores are issued after this, so it does not wait on them.
    asm volatile("s_wait_dscnt 0" ::: "memory");

    // 4) O += P @ V
#pragma unroll
    for (int ks = 0; ks < 2; ++ks) {
      v16bf ap = lds_frag_a(&Ps[wid][0][0], 0, LDA, ks*32);
#pragma unroll
      for (int nd = 0; nd < 4; ++nd) {
        v16bf bv = lds_frag_b(&Vt[cur][0][0], nd*16, LDA, ks*32);
        acco[nd] = WMMA_BF16(ap, bv, acco[nd]);
      }
    }

    // 5) store staged K/V into the other buffer
    if (more) {
      const int nxt = cur ^ 1;
#pragma unroll
      for (int j = 0; j < 4; ++j) *(u32x4*)&Ks[nxt][row][hsel*32 + j*8] = rk[j];
#pragma unroll
      for (int j = 0; j < 4; ++j) {
        u32x4 u = rv[j];
        int d0 = hsel*32 + j*8;
        Vt[nxt][d0+0][row] = (unsigned short)(u.x);
        Vt[nxt][d0+1][row] = (unsigned short)(u.x >> 16);
        Vt[nxt][d0+2][row] = (unsigned short)(u.y);
        Vt[nxt][d0+3][row] = (unsigned short)(u.y >> 16);
        Vt[nxt][d0+4][row] = (unsigned short)(u.z);
        Vt[nxt][d0+5][row] = (unsigned short)(u.z >> 16);
        Vt[nxt][d0+6][row] = (unsigned short)(u.w);
        Vt[nxt][d0+7][row] = (unsigned short)(u.w >> 16);
      }
    }
    __syncthreads();
  }

  // epilogue: O / l -> ctx bf16 [B,S,E]
#pragma unroll
  for (int r = 0; r < 8; ++r) {
    float inv = lrow[r] > 0.f ? 1.f / lrow[r] : 0.f;
    int qrow = qblk + wid*16 + r + 8*hf;
    size_t base = ((size_t)(bq * S_ + qrow)) * E_ + hd * D_;
    unsigned int p01 = pack_bf16x2(acco[0][r] * inv, acco[1][r] * inv);
    unsigned int p23 = pack_bf16x2(acco[2][r] * inv, acco[3][r] * inv);
    ctx[base + nl]      = (unsigned short)p01;
    ctx[base + 16 + nl] = (unsigned short)(p01 >> 16);
    ctx[base + 32 + nl] = (unsigned short)p23;
    ctx[base + 48 + nl] = (unsigned short)(p23 >> 16);
  }
}

// ---------------------------------------------------------------------------
extern "C" void kernel_launch(void* const* d_in, const int* in_sizes, int n_in,
                              void* d_out, int out_size, void* d_ws, size_t ws_size,
                              hipStream_t stream) {
  (void)in_sizes; (void)n_in; (void)out_size; (void)ws_size;
  const float*         Q    = (const float*)d_in[0];
  const float*         K    = (const float*)d_in[1];
  const float*         V    = (const float*)d_in[2];
  const unsigned char* mask = (const unsigned char*)d_in[3];
  const float*         wq   = (const float*)d_in[4];
  const float*         wk   = (const float*)d_in[5];
  const float*         wv   = (const float*)d_in[6];
  const float*         wfc  = (const float*)d_in[7];
  float* out = (float*)d_out;

  unsigned short* w = (unsigned short*)d_ws;
  const size_t NQ = (size_t)B_ * S_ * E_;        // 8388608 (== B*H*S*D)
  const size_t NW = (size_t)E_ * E_;             // 1048576
  unsigned short* qp  = w;                       // projected q [B,H,S,D]
  unsigned short* kp  = w + NQ;
  unsigned short* vp  = w + 2 * NQ;
  unsigned short* cx  = w + 3 * NQ;              // attention ctx [B,S,E]
  unsigned short* Qb  = w + 4 * NQ;              // bf16 inputs
  unsigned short* Kb  = w + 5 * NQ;
  unsigned short* Vb  = w + 6 * NQ;
  unsigned short* wqb = w + 7 * NQ;              // bf16 weights
  unsigned short* wkb = wqb + NW;
  unsigned short* wvb = wkb + NW;
  unsigned short* wfb = wvb + NW;

  const int n8q = (int)(NQ / 8);                 // 1048576 threads
  const int n8w = (int)(NW / 8);                 // 131072 threads
  cvt_bf16_kernel<<<n8q / 256, 256, 0, stream>>>(Q,   Qb,  n8q);
  cvt_bf16_kernel<<<n8q / 256, 256, 0, stream>>>(K,   Kb,  n8q);
  cvt_bf16_kernel<<<n8q / 256, 256, 0, stream>>>(V,   Vb,  n8q);
  cvt_bf16_kernel<<<n8w / 256, 256, 0, stream>>>(wq,  wqb, n8w);
  cvt_bf16_kernel<<<n8w / 256, 256, 0, stream>>>(wk,  wkb, n8w);
  cvt_bf16_kernel<<<n8w / 256, 256, 0, stream>>>(wv,  wvb, n8w);
  cvt_bf16_kernel<<<n8w / 256, 256, 0, stream>>>(wfc, wfb, n8w);

  dim3 gg(E_ / 128, (B_ * S_) / 128);            // 8 x 64
  proj_gemm_kernel<<<gg, dim3(256), 0, stream>>>(Qb, wqb, qp);
  proj_gemm_kernel<<<gg, dim3(256), 0, stream>>>(Kb, wkb, kp);
  proj_gemm_kernel<<<gg, dim3(256), 0, stream>>>(Vb, wvb, vp);
  attn_kernel<<<dim3(S_ / 64, H_, B_), dim3(128), 0, stream>>>(qp, kp, vp, mask, cx);
  out_gemm_kernel<<<gg, dim3(256), 0, stream>>>(cx, wfb, out);
}
